// XMP_66778151518650
// MI455X (gfx1250) — matrix operand order repaired and checked by reading, not compiled, not run on hardware
//
#include <hip/hip_runtime.h>
#include <hip/hip_bf16.h>

// ---------------------------------------------------------------------------
// CDNA5 (gfx1250) full forward of the dual-branch performer.
// All matmuls on v_wmma_f32_16x16x32_bf16: wave32, 4 waves/block, 64x64x32
// block tile (4 WMMAs per wave per K-step), double-buffered LDS staging with
// vectorized b128 global loads + branchless clamped edge path, global
// prefetch of future K-tiles.
// ---------------------------------------------------------------------------

typedef __attribute__((ext_vector_type(16))) __bf16 v16bf;
typedef __attribute__((ext_vector_type(8)))  __bf16 v8bf;
typedef __attribute__((ext_vector_type(8)))  float  v8f;

#define B4     4
#define SEQL   8192
#define EMBD   128
#define NHEAD  8
#define DHEAD  64
#define DINNER 512
#define NBF    266     // int(64*ln(64))
#define NBFP   272     // padded leading dim (multiple of 8 -> 16B rows)

__device__ __forceinline__ float bf2f(__bf16 x) {
  unsigned short u = __builtin_bit_cast(unsigned short, x);
  unsigned v = (unsigned)u << 16;
  return __builtin_bit_cast(float, v);
}
__device__ __forceinline__ __bf16 f2bf(float f) {
  unsigned u = __builtin_bit_cast(unsigned, f);
  unsigned r = (u + 0x7FFFu + ((u >> 16) & 1u)) >> 16;
  return __builtin_bit_cast(__bf16, (unsigned short)r);
}

// ------------------------------- GEMM --------------------------------------

enum { GF_TRANSA = 1, GF_TRANSB = 2, GF_GELU = 4, GF_RELU = 8 };

struct GemmP {
  const __bf16* A;  long sAb, sAh; int lda;
  const __bf16* Bm; long sBb, sBh; int ldb;
  const float* bias;                      // length N (shared across batch)
  const float* resid; long sRb, sRh; int ldr;
  const float* rowscale; long sSb, sSh;   // per-row multiplier
  float*  C;   long sCb,  sCh;  int ldc;
  __bf16* Cbf; long sCfb, sCfh; int ldcf;
  float cscale;
  int M, N, K, H, flags;
};

__global__ __launch_bounds__(128) void wmma_gemm_bf16(GemmP p) {
  __shared__ __attribute__((aligned(16))) __bf16 As[2][64][40];
  __shared__ __attribute__((aligned(16))) __bf16 Bt[2][64][40];  // stored [n][k]

  const int t    = threadIdx.x;
  const int lane = t & 31, wave = t >> 5;
  const int m0 = blockIdx.y * 64, n0 = blockIdx.x * 64;
  const int z  = blockIdx.z;
  const int zb = z / p.H, zh = z - zb * p.H;
  const __bf16* Ap = p.A  + zb * p.sAb + zh * p.sAh;
  const __bf16* Bp = p.Bm + zb * p.sBb + zh * p.sBh;

  const int wr = (wave >> 1) * 32, wc = (wave & 1) * 32;
  const int lm = lane & 15, kh = lane >> 4;

  v8f acc[2][2] = {};
  const int kt = (p.K + 31) >> 5;
  const __bf16 zbf = __builtin_bit_cast(__bf16, (unsigned short)0);

  auto load_tile = [&](int ti, int buf) {
    const int k0 = ti * 32;
    const bool fullK = (k0 + 32 <= p.K);
    // ---------------- A tile: 64 x 32 -> As[m][k] ----------------
    if (!(p.flags & GF_TRANSA)) {
      if (fullK && (m0 + 64 <= p.M)) {
        int r = t >> 2, c0 = (t & 3) * 8;
        #pragma unroll
        for (int hh = 0; hh < 2; ++hh) {
          int rr = r + hh * 32;
          v8bf v = *(const v8bf*)(Ap + (long)(m0 + rr) * p.lda + k0 + c0);
          *(v8bf*)&As[buf][rr][c0] = v;
        }
      } else {
        int r = t >> 2, c0 = (t & 3) * 8;
        #pragma unroll
        for (int hh = 0; hh < 2; ++hh) {
          int rr = r + hh * 32, gm = m0 + rr;
          int cm = gm < p.M ? gm : p.M - 1;
          #pragma unroll
          for (int j = 0; j < 8; ++j) {
            int gk = k0 + c0 + j;
            int ck = gk < p.K ? gk : p.K - 1;
            __bf16 v = Ap[(long)cm * p.lda + ck];
            if (gm >= p.M || gk >= p.K) v = zbf;
            As[buf][rr][c0 + j] = v;
          }
        }
      }
    } else {  // A stored [K, M]; contiguous along M
      if (fullK && (m0 + 64 <= p.M)) {
        int k = t >> 2, mc = (t & 3) * 8;
        #pragma unroll
        for (int hh = 0; hh < 2; ++hh) {
          int mm = mc + hh * 32;
          v8bf v = *(const v8bf*)(Ap + (long)(k0 + k) * p.lda + m0 + mm);
          #pragma unroll
          for (int j = 0; j < 8; ++j) As[buf][mm + j][k] = v[j];
        }
      } else {
        int k = t >> 2, mc = (t & 3) * 8;
        int gk = k0 + k, ck = gk < p.K ? gk : p.K - 1;
        #pragma unroll
        for (int hh = 0; hh < 2; ++hh) {
          #pragma unroll
          for (int j = 0; j < 8; ++j) {
            int mm = mc + hh * 32 + j, gm = m0 + mm;
            int cm = gm < p.M ? gm : p.M - 1;
            __bf16 v = Ap[(long)ck * p.lda + cm];
            if (gm >= p.M || gk >= p.K) v = zbf;
            As[buf][mm][k] = v;
          }
        }
      }
    }
    // ---------------- B tile: 32 x 64 -> Bt[n][k] ----------------
    if (!(p.flags & GF_TRANSB)) {  // B stored [K, N]; contiguous along N
      if (fullK && (n0 + 64 <= p.N)) {
        int k = t >> 2, nc = (t & 3) * 8;
        #pragma unroll
        for (int hh = 0; hh < 2; ++hh) {
          int nn = nc + hh * 32;
          v8bf v = *(const v8bf*)(Bp + (long)(k0 + k) * p.ldb + n0 + nn);
          #pragma unroll
          for (int j = 0; j < 8; ++j) Bt[buf][nn + j][k] = v[j];
        }
      } else {
        int k = t >> 2, nc = (t & 3) * 8;
        int gk = k0 + k, ck = gk < p.K ? gk : p.K - 1;
        #pragma unroll
        for (int hh = 0; hh < 2; ++hh) {
          #pragma unroll
          for (int j = 0; j < 8; ++j) {
            int nn = nc + hh * 32 + j, gn = n0 + nn;
            int cn = gn < p.N ? gn : p.N - 1;
            __bf16 v = Bp[(long)ck * p.ldb + cn];
            if (gn >= p.N || gk >= p.K) v = zbf;
            Bt[buf][nn][k] = v;
          }
        }
      }
    } else {  // B stored [N, K]; contiguous along K
      if (fullK && (n0 + 64 <= p.N)) {
        int n = t >> 2, kc = (t & 3) * 8;
        #pragma unroll
        for (int hh = 0; hh < 2; ++hh) {
          int nn = n + hh * 32;
          v8bf v = *(const v8bf*)(Bp + (long)(n0 + nn) * p.ldb + k0 + kc);
          *(v8bf*)&Bt[buf][nn][kc] = v;
        }
      } else {
        int n = t >> 2, kc = (t & 3) * 8;
        #pragma unroll
        for (int hh = 0; hh < 2; ++hh) {
          int nn = n + hh * 32, gn = n0 + nn;
          int cn = gn < p.N ? gn : p.N - 1;
          #pragma unroll
          for (int j = 0; j < 8; ++j) {
            int gk = k0 + kc + j;
            int ck = gk < p.K ? gk : p.K - 1;
            __bf16 v = Bp[(long)cn * p.ldb + ck];
            if (gn >= p.N || gk >= p.K) v = zbf;
            Bt[buf][nn][kc + j] = v;
          }
        }
      }
    }
  };

  load_tile(0, 0);
  __syncthreads();

  for (int ti = 0; ti < kt; ++ti) {
    int buf = ti & 1;
    if (ti + 1 < kt) load_tile(ti + 1, buf ^ 1);
    if ((ti + 2 < kt) && !(p.flags & GF_TRANSA)) {
      int gm = m0 + (t & 63);
      if (gm < p.M) __builtin_prefetch(Ap + (long)gm * p.lda + (ti + 2) * 32, 0, 1);
    }
    // Fragments per ISA 16-bit layouts (A 16x32: lane=M, kh selects K-half;
    // B from transposed LDS: lane=N, same K split as A).
    const int ra = wr + lm, rb = wc + lm;
    v16bf a[2], b[2];
    #pragma unroll
    for (int i = 0; i < 2; ++i) {
      v8bf lo = *(const v8bf*)&As[buf][ra + i * 16][kh * 8];
      v8bf hi = *(const v8bf*)&As[buf][ra + i * 16][16 + kh * 8];
      a[i] = __builtin_shufflevector(lo, hi, 0,1,2,3,4,5,6,7,8,9,10,11,12,13,14,15);
      v8bf bl = *(const v8bf*)&Bt[buf][rb + i * 16][kh * 8];
      v8bf bh = *(const v8bf*)&Bt[buf][rb + i * 16][16 + kh * 8];
      b[i] = __builtin_shufflevector(bl, bh, 0,1,2,3,4,5,6,7,8,9,10,11,12,13,14,15);
    }
    #pragma unroll
    for (int i = 0; i < 2; ++i)
      #pragma unroll
      for (int j = 0; j < 2; ++j)
        acc[i][j] = __builtin_amdgcn_wmma_f32_16x16x32_bf16(
            false, a[i], false, b[j], (short)0, acc[i][j], false, false);
    __syncthreads();
  }

  // Epilogue: C row m = r + (lane>>4)*8, col n = lane&15 (ISA C/D layout)
  #pragma unroll
  for (int i = 0; i < 2; ++i) {
    #pragma unroll
    for (int j = 0; j < 2; ++j) {
      const int nn = n0 + wc + j * 16 + lm;
      const int mb = m0 + wr + i * 16 + kh * 8;
      #pragma unroll
      for (int r = 0; r < 8; ++r) {
        int m = mb + r;
        if (m < p.M && nn < p.N) {
          float v = acc[i][j][r];
          if (p.bias) v += p.bias[nn];
          if (p.flags & GF_GELU) {
            float x = v;
            v = 0.5f * x * (1.f + tanhf(0.7978845608f * (x + 0.044715f * x * x * x)));
          }
          if (p.flags & GF_RELU) v = v > 0.f ? v : 0.f;
          v *= p.cscale;
          if (p.rowscale) v *= p.rowscale[zb * p.sSb + zh * p.sSh + m];
          if (p.resid)    v += p.resid[zb * p.sRb + zh * p.sRh + (long)m * p.ldr + nn];
          if (p.C)   p.C  [zb * p.sCb  + zh * p.sCh  + (long)m * p.ldc  + nn] = v;
          if (p.Cbf) p.Cbf[zb * p.sCfb + zh * p.sCfh + (long)m * p.ldcf + nn] = f2bf(v);
        }
      }
    }
  }
}

// --------------------------- small kernels ---------------------------------

__global__ void k_gather_embed(const int* series, const float* emb, float* x, int total) {
  int i = blockIdx.x * blockDim.x + threadIdx.x;
  if (i >= total) return;
  int c = i & (EMBD - 1);
  int s = i / EMBD;
  x[i] = emb[series[s] * EMBD + c];
}

__global__ void k_conv_embed(const float* x, const float* w, const float* bias,
                             const float* pos, const float* cls, float* out,
                             int D, int ksz, int pool, int Ntok) {
  int i = blockIdx.x * blockDim.x + threadIdx.x;
  int total = B4 * (Ntok + 1) * D;
  if (i >= total) return;
  int d = i % D, pn = (i / D) % (Ntok + 1), b = i / (D * (Ntok + 1));
  float val;
  if (pn == 0) {
    val = cls[d];
  } else {
    int p = pn - 1, l0 = p * pool, pad = (ksz - 1) / 2;
    float mx = -3.4e38f;
    for (int l = 0; l < pool; ++l) {
      float acc = bias[d];
      for (int tt = 0; tt < ksz; ++tt) {
        int sl = l0 + l + tt - pad;
        if (sl >= 0 && sl < SEQL) {
          const float* xp = x + ((long)b * SEQL + sl) * EMBD;
          const float* wp = w + (long)d * EMBD * ksz + tt;
          for (int c = 0; c < EMBD; ++c) acc += xp[c] * wp[c * ksz];
        }
      }
      acc = acc > 0.f ? acc : 0.f;
      mx = acc > mx ? acc : mx;
    }
    val = mx;
  }
  out[i] = val + pos[pn * D + d];
}

__global__ __launch_bounds__(256) void k_layernorm(const float* src, long rowStride,
    const float* g, const float* bta, int rows, int n,
    __bf16* out_bf, long obStride, float* out_f, long ofStride) {
  int row = blockIdx.x;
  if (row >= rows) return;
  const float* p = src + (long)row * rowStride;
  __shared__ float red[256];
  int tid = threadIdx.x;
  float s = 0.f;
  for (int i = tid; i < n; i += 256) s += p[i];
  red[tid] = s; __syncthreads();
  for (int o = 128; o > 0; o >>= 1) { if (tid < o) red[tid] += red[tid + o]; __syncthreads(); }
  float mean = red[0] / n; __syncthreads();
  float v = 0.f;
  for (int i = tid; i < n; i += 256) { float dd = p[i] - mean; v += dd * dd; }
  red[tid] = v; __syncthreads();
  for (int o = 128; o > 0; o >>= 1) { if (tid < o) red[tid] += red[tid + o]; __syncthreads(); }
  float inv = rsqrtf(red[0] / n + 1e-5f);
  for (int i = tid; i < n; i += 256) {
    float o = (p[i] - mean) * inv * g[i] + bta[i];
    if (out_bf) out_bf[(long)row * obStride + i] = f2bf(o);
    if (out_f)  out_f [(long)row * ofStride + i] = o;
  }
}

__global__ void k_rows_bf16(const float* src, long rowStride, int rows, int n, __bf16* dst) {
  int i = blockIdx.x * blockDim.x + threadIdx.x;
  if (i >= rows * n) return;
  int r = i / n, c = i - r * n;
  dst[i] = f2bf(src[(long)r * rowStride + c]);
}

__global__ void k_diag(const __bf16* qbf, float* diag, int N) {
  int i = blockIdx.x * blockDim.x + threadIdx.x;
  if (i >= B4 * NHEAD * N) return;
  int n = i % N, h = (i / N) % NHEAD, b = i / (N * NHEAD);
  const __bf16* p = qbf + ((long)(b * N + n)) * DINNER + h * DHEAD;
  float s = 0.f;
  for (int d = 0; d < DHEAD; ++d) { float f = bf2f(p[d]); s += f * f; }
  diag[i] = 0.5f * s;
}

__global__ void k_rowmax(const float* dd, float* rmax, int rows, int F, int ld) {
  int i = blockIdx.x * blockDim.x + threadIdx.x;
  if (i >= rows) return;
  const float* p = dd + (long)i * ld;
  float m = -3.4e38f;
  for (int f = 0; f < F; ++f) m = fmaxf(m, p[f]);
  rmax[i] = m;
}

__global__ __launch_bounds__(256) void k_gmax(const float* dd, long rows, int F, int ld,
                                              float* out) {
  __shared__ float red[256];
  int tid = threadIdx.x;
  float m = -3.4e38f;
  for (long r = tid; r < rows; r += 256) {
    const float* p = dd + r * ld;
    for (int f = 0; f < F; ++f) m = fmaxf(m, p[f]);
  }
  red[tid] = m; __syncthreads();
  for (int o = 128; o > 0; o >>= 1) { if (tid < o) red[tid] = fmaxf(red[tid], red[tid + o]); __syncthreads(); }
  if (tid == 0) out[0] = red[0];
}

__global__ void k_featexp(const float* dd, const float* diag, const float* rowm,
                          const float* gmax, __bf16* out, long rows, int F, int ld,
                          float ratio) {
  long i = (long)blockIdx.x * blockDim.x + threadIdx.x;
  if (i >= rows * F) return;
  long row = i / F;
  int  f   = (int)(i - row * F);
  float m = rowm ? rowm[row] : gmax[0];
  out[row * ld + f] = f2bf(ratio * (expf(dd[row * ld + f] - diag[row] - m) + 1e-4f));
}

__global__ void k_colsum(const __bf16* kp, float* ksum, int Z, int N, int ld) {
  int i = blockIdx.x * blockDim.x + threadIdx.x;
  if (i >= Z * NBF) return;
  int z = i / NBF, f = i - z * NBF;
  const __bf16* p = kp + (long)z * N * ld + f;
  float s = 0.f;
  for (int n = 0; n < N; ++n) s += bf2f(p[(long)n * ld]);
  ksum[i] = s;
}

__global__ void k_dinv(const __bf16* qp, const float* ksum, float* dinv, int Z, int N, int ld) {
  int i = blockIdx.x * blockDim.x + threadIdx.x;
  if (i >= Z * N) return;
  int z = i / N, n = i - z * N;
  const __bf16* p = qp + ((long)z * N + n) * ld;
  const float* ks = ksum + (long)z * NBF;
  float s = 0.f;
  for (int f = 0; f < NBF; ++f) s += bf2f(p[f]) * ks[f];
  dinv[i] = 1.f / s;
}

__global__ void k_concat_ctx(const __bf16* h, const float* patTok, long patRowStride,
                             int Nc, int dout, __bf16* dst) {
  int i = blockIdx.x * blockDim.x + threadIdx.x;
  int Nk = 1 + Nc;
  if (i >= B4 * Nk * dout) return;
  int c = i % dout, tk = (i / dout) % Nk, b = i / (dout * Nk);
  if (tk == 0) dst[i] = h[b * dout + c];
  else         dst[i] = f2bf(patTok[(long)b * patRowStride + (long)tk * dout + c]);
}

__global__ __launch_bounds__(128) void k_cross_attn(const float* q, const float* kv,
                                                    __bf16* out, int Nk) {
  int b = blockIdx.x >> 3, h = blockIdx.x & 7;
  __shared__ float sc[520];
  __shared__ float inv_s;
  const float* qh  = q  + (long)b * DINNER + h * DHEAD;
  const float* kvb = kv + (long)b * Nk * (2 * DINNER);
  for (int tk = threadIdx.x; tk < Nk; tk += blockDim.x) {
    const float* kp = kvb + (long)tk * (2 * DINNER) + h * DHEAD;
    float s = 0.f;
    for (int d = 0; d < DHEAD; ++d) s += qh[d] * kp[d];
    sc[tk] = s * 0.125f;                 // DH^-0.5
  }
  __syncthreads();
  if (threadIdx.x == 0) {
    float m = -3.4e38f;
    for (int i = 0; i < Nk; ++i) m = fmaxf(m, sc[i]);
    float s = 0.f;
    for (int i = 0; i < Nk; ++i) { float e = expf(sc[i] - m); sc[i] = e; s += e; }
    inv_s = 1.f / s;
  }
  __syncthreads();
  float inv = inv_s;
  for (int d = threadIdx.x; d < DHEAD; d += blockDim.x) {
    float acc = 0.f;
    for (int tk = 0; tk < Nk; ++tk)
      acc += sc[tk] * kvb[(long)tk * (2 * DINNER) + DINNER + h * DHEAD + d];
    out[(long)b * DINNER + h * DHEAD + d] = f2bf(acc * inv);
  }
}

__global__ void k_logits(const float* lnrow, const float* w, const float* bias,
                         float* out, int D, int add) {
  int b = blockIdx.x, c = threadIdx.x;
  if (c >= 10) return;
  const float* p = lnrow + (long)b * D;
  float s = bias[c];
  for (int d = 0; d < D; ++d) s += p[d] * w[d * 10 + c];
  if (add) out[b * 10 + c] += s; else out[b * 10 + c] = s;
}

// ------------------------------ host side ----------------------------------

struct PerfW {
  const float *ln1_g, *ln1_b, *bq, *bk, *bv, *bo, *ln2_g, *ln2_b, *b1, *b2;
  const __bf16 *wq, *wk, *wv, *wo, *proj, *w1, *w2;
};
struct CrossW {
  const float *pi_b, *po_b, *ln_g, *ln_b, *bo;
  const __bf16 *pi_w, *po_w, *wq, *wkv, *wo;
};
struct Scratch {
  __bf16 *hbf, *qbf, *kbf, *vbf, *qpbf, *kpbf, *ctxbf, *attnbf, *ffbf;
  float  *dd, *diag, *rmax, *ksum, *dinv, *ctxf, *gmax;
  __bf16 *clsbf, *hbf2, *ctxcat, *attnbf2, *obf;
  float  *xbuf, *kv, *qbuf, *lnhead;
};

static inline void launch_gemm(hipStream_t st, const GemmP& p, int batch) {
  dim3 g((unsigned)((p.N + 63) / 64), (unsigned)((p.M + 63) / 64), (unsigned)batch);
  wmma_gemm_bf16<<<g, dim3(128), 0, st>>>(p);
}
static inline int blks(long n) { return (int)((n + 255) / 256); }

static void run_performer(hipStream_t st, float* tok, int N, int D,
                          const PerfW& w, const Scratch& S) {
  const int rows = B4 * N, Z = B4 * NHEAD;
  const float DN = 0.35355339059f;                 // 64^-0.25
  const float RATIO = 0.06131393394f;              // 266^-0.5
  k_layernorm<<<rows, 256, 0, st>>>(tok, D, w.ln1_g, w.ln1_b, rows, D,
                                    S.hbf, D, nullptr, 0);
  GemmP p{};
  p.A = S.hbf; p.lda = D; p.Bm = w.wq; p.ldb = DINNER; p.bias = w.bq;
  p.Cbf = S.qbf; p.ldcf = DINNER; p.M = rows; p.N = DINNER; p.K = D;
  p.H = 1; p.cscale = DN;
  launch_gemm(st, p, 1);
  p.Bm = w.wk; p.bias = w.bk; p.Cbf = S.kbf;                 launch_gemm(st, p, 1);
  p.Bm = w.wv; p.bias = w.bv; p.Cbf = S.vbf; p.cscale = 1.f; launch_gemm(st, p, 1);

  // dd = (q*dn) @ proj^T  (batched over B*H), then FAVOR+ features (query)
  GemmP f{};
  f.A = S.qbf; f.lda = DINNER; f.sAb = (long)N * DINNER; f.sAh = DHEAD;
  f.Bm = w.proj; f.ldb = DHEAD; f.flags = GF_TRANSB;
  f.C = S.dd; f.ldc = NBFP; f.sCb = (long)NHEAD * N * NBFP; f.sCh = (long)N * NBFP;
  f.M = N; f.N = NBF; f.K = DHEAD; f.H = NHEAD; f.cscale = 1.f;
  launch_gemm(st, f, Z);
  k_diag<<<blks((long)Z * N), 256, 0, st>>>(S.qbf, S.diag, N);
  k_rowmax<<<blks((long)Z * N), 256, 0, st>>>(S.dd, S.rmax, Z * N, NBF, NBFP);
  k_featexp<<<blks((long)Z * N * NBF), 256, 0, st>>>(S.dd, S.diag, S.rmax, nullptr,
      S.qpbf, (long)Z * N, NBF, NBFP, RATIO);
  // key features (global max)
  f.A = S.kbf; launch_gemm(st, f, Z);
  k_diag<<<blks((long)Z * N), 256, 0, st>>>(S.kbf, S.diag, N);
  k_gmax<<<1, 256, 0, st>>>(S.dd, (long)Z * N, NBF, NBFP, S.gmax);
  k_featexp<<<blks((long)Z * N * NBF), 256, 0, st>>>(S.dd, S.diag, nullptr, S.gmax,
      S.kpbf, (long)Z * N, NBF, NBFP, RATIO);
  k_colsum<<<blks((long)Z * NBF), 256, 0, st>>>(S.kpbf, S.ksum, Z, N, NBFP);
  k_dinv<<<blks((long)Z * N), 256, 0, st>>>(S.qpbf, S.ksum, S.dinv, Z, N, NBFP);

  // ctx = kp^T @ v  (M=266, K=N, N=64)
  GemmP c{};
  c.A = S.kpbf; c.lda = NBFP; c.flags = GF_TRANSA;
  c.sAb = (long)NHEAD * N * NBFP; c.sAh = (long)N * NBFP;
  c.Bm = S.vbf; c.ldb = DINNER; c.sBb = (long)N * DINNER; c.sBh = DHEAD;
  c.C = S.ctxf; c.ldc = DHEAD; c.sCb = (long)NHEAD * NBF * DHEAD; c.sCh = (long)NBF * DHEAD;
  c.Cbf = S.ctxbf; c.ldcf = DHEAD; c.sCfb = c.sCb; c.sCfh = c.sCh;
  c.M = NBF; c.N = DHEAD; c.K = N; c.H = NHEAD; c.cscale = 1.f;
  launch_gemm(st, c, Z);

  // out = (qp @ ctx) * dinv, merged into [B*N, 512]
  GemmP o{};
  o.A = S.qpbf; o.lda = NBFP; o.sAb = (long)NHEAD * N * NBFP; o.sAh = (long)N * NBFP;
  o.Bm = S.ctxbf; o.ldb = DHEAD; o.sBb = (long)NHEAD * NBF * DHEAD; o.sBh = (long)NBF * DHEAD;
  o.rowscale = S.dinv; o.sSb = (long)NHEAD * N; o.sSh = N;
  o.Cbf = S.attnbf; o.ldcf = DINNER; o.sCfb = (long)N * DINNER; o.sCfh = DHEAD;
  o.M = N; o.N = DHEAD; o.K = NBF; o.H = NHEAD; o.cscale = 1.f;
  launch_gemm(st, o, Z);

  // x += attn @ wo + bo
  GemmP wo{};
  wo.A = S.attnbf; wo.lda = DINNER; wo.Bm = w.wo; wo.ldb = D; wo.bias = w.bo;
  wo.resid = tok; wo.ldr = D; wo.C = tok; wo.ldc = D;
  wo.M = rows; wo.N = D; wo.K = DINNER; wo.H = 1; wo.cscale = 1.f;
  launch_gemm(st, wo, 1);

  // MLP
  k_layernorm<<<rows, 256, 0, st>>>(tok, D, w.ln2_g, w.ln2_b, rows, D,
                                    S.hbf, D, nullptr, 0);
  GemmP m1{};
  m1.A = S.hbf; m1.lda = D; m1.Bm = w.w1; m1.ldb = 4 * D; m1.bias = w.b1;
  m1.flags = GF_GELU; m1.Cbf = S.ffbf; m1.ldcf = 4 * D;
  m1.M = rows; m1.N = 4 * D; m1.K = D; m1.H = 1; m1.cscale = 1.f;
  launch_gemm(st, m1, 1);
  GemmP m2{};
  m2.A = S.ffbf; m2.lda = 4 * D; m2.Bm = w.w2; m2.ldb = D; m2.bias = w.b2;
  m2.resid = tok; m2.ldr = D; m2.C = tok; m2.ldc = D;
  m2.M = rows; m2.N = D; m2.K = 4 * D; m2.H = 1; m2.cscale = 1.f;
  launch_gemm(st, m2, 1);
}

static void run_cross(hipStream_t st, float* tokSrc, int Nsrc, int din,
                      const float* tokCtx, int Nc, int dout,
                      const CrossW& w, const Scratch& S) {
  const int Nk = 1 + Nc;
  k_rows_bf16<<<blks((long)B4 * din), 256, 0, st>>>(tokSrc, (long)(Nsrc + 1) * din,
                                                    B4, din, S.clsbf);
  GemmP p{};
  p.A = S.clsbf; p.lda = din; p.Bm = w.pi_w; p.ldb = dout; p.bias = w.pi_b;
  p.C = S.xbuf; p.ldc = dout; p.M = B4; p.N = dout; p.K = din; p.H = 1; p.cscale = 1.f;
  launch_gemm(st, p, 1);
  k_layernorm<<<B4, 256, 0, st>>>(S.xbuf, dout, w.ln_g, w.ln_b, B4, dout,
                                  S.hbf2, dout, nullptr, 0);
  k_concat_ctx<<<blks((long)B4 * Nk * dout), 256, 0, st>>>(S.hbf2, tokCtx,
      (long)(Nc + 1) * dout, Nc, dout, S.ctxcat);
  GemmP kvp{};
  kvp.A = S.ctxcat; kvp.lda = dout; kvp.Bm = w.wkv; kvp.ldb = 2 * DINNER;
  kvp.C = S.kv; kvp.ldc = 2 * DINNER;
  kvp.M = B4 * Nk; kvp.N = 2 * DINNER; kvp.K = dout; kvp.H = 1; kvp.cscale = 1.f;
  launch_gemm(st, kvp, 1);
  GemmP qp{};
  qp.A = S.hbf2; qp.lda = dout; qp.Bm = w.wq; qp.ldb = DINNER;
  qp.C = S.qbuf; qp.ldc = DINNER; qp.M = B4; qp.N = DINNER; qp.K = dout;
  qp.H = 1; qp.cscale = 1.f;
  launch_gemm(st, qp, 1);
  k_cross_attn<<<B4 * NHEAD, 128, 0, st>>>(S.qbuf, S.kv, S.attnbf2, Nk);
  GemmP op{};
  op.A = S.attnbf2; op.lda = DINNER; op.Bm = w.wo; op.ldb = dout; op.bias = w.bo;
  op.Cbf = S.obf; op.ldcf = dout; op.M = B4; op.N = dout; op.K = DINNER;
  op.H = 1; op.cscale = 1.f;
  launch_gemm(st, op, 1);
  GemmP pp{};
  pp.A = S.obf; pp.lda = dout; pp.Bm = w.po_w; pp.ldb = din; pp.bias = w.po_b;
  pp.resid = tokSrc; pp.ldr = (Nsrc + 1) * din;
  pp.C = tokSrc; pp.ldc = (Nsrc + 1) * din;
  pp.M = B4; pp.N = din; pp.K = dout; pp.H = 1; pp.cscale = 1.f;
  launch_gemm(st, pp, 1);
}

extern "C" void kernel_launch(void* const* d_in, const int* in_sizes, int n_in,
                              void* d_out, int out_size, void* d_ws, size_t ws_size,
                              hipStream_t stream) {
  (void)in_sizes; (void)n_in; (void)out_size; (void)ws_size;
  char* wp = (char*)d_ws;
  auto alloc = [&](size_t bytes) -> void* {
    void* p = wp; wp += (bytes + 255) & ~(size_t)255; return p;
  };
  int cur = 0;
  auto fp = [&]() { return (const float*)d_in[cur++]; };
  auto cvt = [&](const float* src, int count) -> const __bf16* {
    __bf16* dst = (__bf16*)alloc((size_t)count * 2);
    k_rows_bf16<<<blks(count), 256, 0, stream>>>(src, count, 1, count, dst);
    return dst;
  };

  const int* series = (const int*)d_in[cur++];
  const float* embw = fp();
  const float* smE[4]; for (int i = 0; i < 4; ++i) smE[i] = fp();
  const float* lgE[4]; for (int i = 0; i < 4; ++i) lgE[i] = fp();

  auto fillPerf = [&](PerfW& w, int D) {
    w.ln1_g = fp(); w.ln1_b = fp();
    const float* wq = fp(); w.bq = fp();
    const float* wk = fp(); w.bk = fp();
    const float* wv = fp(); w.bv = fp();
    const float* wo = fp(); w.bo = fp();
    const float* pj = fp();
    w.ln2_g = fp(); w.ln2_b = fp();
    const float* w1 = fp(); w.b1 = fp();
    const float* w2 = fp(); w.b2 = fp();
    w.wq = cvt(wq, D * DINNER); w.wk = cvt(wk, D * DINNER); w.wv = cvt(wv, D * DINNER);
    w.wo = cvt(wo, DINNER * D); w.proj = cvt(pj, NBF * DHEAD);
    w.w1 = cvt(w1, D * 4 * D);  w.w2 = cvt(w2, 4 * D * D);
  };
  auto fillCross = [&](CrossW& w, int din, int dout) {
    const float* pi = fp(); w.pi_b = fp();
    const float* po = fp(); w.po_b = fp();
    w.ln_g = fp(); w.ln_b = fp();
    const float* wq = fp(); const float* wkv = fp(); const float* wo = fp(); w.bo = fp();
    w.pi_w = cvt(pi, din * dout); w.po_w = cvt(po, dout * din);
    w.wq = cvt(wq, dout * DINNER); w.wkv = cvt(wkv, dout * 2 * DINNER);
    w.wo = cvt(wo, DINNER * dout);
  };

  PerfW smP[3][2], lgP[3][3];
  for (int d = 0; d < 3; ++d) for (int l = 0; l < 2; ++l) fillPerf(smP[d][l], 192);
  for (int d = 0; d < 3; ++d) for (int l = 0; l < 3; ++l) fillPerf(lgP[d][l], 384);
  CrossW crs[3][2][2];
  for (int d = 0; d < 3; ++d)
    for (int c = 0; c < 2; ++c) {
      fillCross(crs[d][c][0], 192, 384);   // sm2lg
      fillCross(crs[d][c][1], 384, 192);   // lg2sm
    }
  const float* smH[4]; for (int i = 0; i < 4; ++i) smH[i] = fp();
  const float* lgH[4]; for (int i = 0; i < 4; ++i) lgH[i] = fp();

  // ---------------- scratch ----------------
  float* xg    = (float*)alloc((size_t)B4 * SEQL * EMBD * 4);
  float* smtok = (float*)alloc((size_t)B4 * 513 * 192 * 4);
  float* lgtok = (float*)alloc((size_t)B4 * 129 * 384 * 4);
  Scratch S{};
  const long maxRD  = (long)B4 * 513 * 192;     // max rows*D
  const long maxRI  = (long)B4 * 513 * DINNER;  // max rows*512
  const long maxFF  = (long)B4 * 513 * 768;     // max rows*4D
  const long maxZNF = (long)32 * 513 * NBFP;
  S.hbf   = (__bf16*)alloc(maxRD * 2);
  S.qbf   = (__bf16*)alloc(maxRI * 2);
  S.kbf   = (__bf16*)alloc(maxRI * 2);
  S.vbf   = (__bf16*)alloc(maxRI * 2);
  S.qpbf  = (__bf16*)alloc(maxZNF * 2);
  S.kpbf  = (__bf16*)alloc(maxZNF * 2);
  S.ctxbf = (__bf16*)alloc((size_t)32 * NBF * DHEAD * 2);
  S.attnbf= (__bf16*)alloc(maxRI * 2);
  S.ffbf  = (__bf16*)alloc(maxFF * 2);
  S.dd    = (float*)alloc(maxZNF * 4);
  S.diag  = (float*)alloc((size_t)32 * 513 * 4);
  S.rmax  = (float*)alloc((size_t)32 * 513 * 4);
  S.ksum  = (float*)alloc((size_t)32 * NBF * 4);
  S.dinv  = (float*)alloc((size_t)32 * 513 * 4);
  S.ctxf  = (float*)alloc((size_t)32 * NBF * DHEAD * 4);
  S.gmax  = (float*)alloc(256);
  S.clsbf = (__bf16*)alloc((size_t)B4 * 384 * 2);
  S.hbf2  = (__bf16*)alloc((size_t)B4 * 384 * 2);
  S.ctxcat= (__bf16*)alloc((size_t)B4 * 513 * 384 * 2);
  S.attnbf2=(__bf16*)alloc((size_t)B4 * DINNER * 2);
  S.obf   = (__bf16*)alloc((size_t)B4 * 384 * 2);
  S.xbuf  = (float*)alloc((size_t)B4 * 384 * 4);
  S.kv    = (float*)alloc((size_t)B4 * 513 * 1024 * 4);
  S.qbuf  = (float*)alloc((size_t)B4 * DINNER * 4);
  S.lnhead= (float*)alloc((size_t)B4 * 384 * 4);

  // ---------------- forward ----------------
  k_gather_embed<<<blks((long)B4 * SEQL * EMBD), 256, 0, stream>>>(
      series, embw, xg, B4 * SEQL * EMBD);
  k_conv_embed<<<blks((long)B4 * 513 * 192), 256, 0, stream>>>(
      xg, smE[0], smE[1], smE[2], smE[3], smtok, 192, 3, 16, 512);
  k_conv_embed<<<blks((long)B4 * 129 * 384), 256, 0, stream>>>(
      xg, lgE[0], lgE[1], lgE[2], lgE[3], lgtok, 384, 7, 64, 128);

  for (int d = 0; d < 3; ++d) {
    for (int l = 0; l < 2; ++l) run_performer(stream, smtok, 513, 192, smP[d][l], S);
    for (int l = 0; l < 3; ++l) run_performer(stream, lgtok, 129, 384, lgP[d][l], S);
    for (int c = 0; c < 2; ++c) {
      run_cross(stream, smtok, 512, 192, lgtok, 128, 384, crs[d][c][0], S);
      run_cross(stream, lgtok, 128, 384, smtok, 512, 192, crs[d][c][1], S);
    }
  }

  float* out = (float*)d_out;
  k_layernorm<<<B4, 256, 0, stream>>>(smtok, (long)513 * 192, smH[0], smH[1],
                                      B4, 192, nullptr, 0, S.lnhead, 192);
  k_logits<<<B4, 16, 0, stream>>>(S.lnhead, smH[2], smH[3], out, 192, 0);
  k_layernorm<<<B4, 256, 0, stream>>>(lgtok, (long)129 * 384, lgH[0], lgH[1],
                                      B4, 384, nullptr, 0, S.lnhead, 384);
  k_logits<<<B4, 16, 0, stream>>>(S.lnhead, lgH[2], lgH[3], out, 384, 1);
}
